// AttentionBlockV3_33165737459773
// MI455X (gfx1250) — matrix operand order repaired and checked by reading, not compiled, not run on hardware
//
#include <hip/hip_runtime.h>
#include <cstdint>

// MI455X (gfx1250) attention block: f16 WMMA, f32 accumulate, flash-attention.
// Kernel 2 uses the CDNA5 Tensor Data Mover (tensor_load_to_lds, TENSORcnt)
// with double-buffered K/V tiles so the DMA of tile kt+1 overlaps the
// softmax+WMMA of tile kt. All WMMA fragments except Q (global b128) are fed by ds_load_b128.
// Scratch layouts (so every staging copy / TDM tile is b128/row friendly):
//   Q: f16 [B,h,N,dh]   (pre-scaled by dh^-0.5)
//   K: f16 [B,h,N,dh]
//   V: f16 [B,h,dh,N]   (transposed -> PV B-fragments contiguous)
//   O: f16 [B,N,C]      (-> proj B-fragments contiguous)
// Scratch use: 4 * 4,194,304 f16 = 33.6 MB of d_ws.

#define BATCH 4
#define CCH   256
#define NSP   4096
#define NHEAD 4
#define DH    64

typedef __attribute__((ext_vector_type(16))) _Float16 v16h;
typedef __attribute__((ext_vector_type(8)))  float    v8f;
typedef __attribute__((ext_vector_type(4)))  unsigned v4u;
typedef __attribute__((ext_vector_type(8)))  unsigned v8u;

union Frag16 { v16h h; unsigned int u[8]; };

__device__ __forceinline__ unsigned short f2h(float f) {
  _Float16 h = (_Float16)f;
  return __builtin_bit_cast(unsigned short, h);
}

__device__ __forceinline__ unsigned pack_f16x2(float a, float b) {
  return (unsigned)f2h(a) | ((unsigned)f2h(b) << 16);  // -> v_cvt_pk_f16_f32
}

__device__ __forceinline__ v8f wmma_f16(const Frag16& a, const Frag16& b, v8f c) {
  // D = A(16x32 f16) * B(32x16 f16) + C(16x16 f32)
  return __builtin_amdgcn_wmma_f32_16x16x32_f16(false, a.h, false, b.h,
                                                (short)0, c, false, false);
}

__device__ __forceinline__ v8f zero8() {
  v8f z;
#pragma unroll
  for (int i = 0; i < 8; ++i) z[i] = 0.0f;
  return z;
}

// --- Tensor Data Mover: 2D tile (64 rows x 64 f16) global -> LDS ------------
// D# per CDNA5 ISA. data_size=2B. LDS row padding:
// pad_interval=4 (32 dwords = 128B of data), pad_amount=3 (4 dwords = 16B)
// -> 144B LDS row stride, matching the [64][72] u16 LDS tiles.
// All inputs are wave-uniform -> descriptor groups live in SGPRs.
__device__ __forceinline__ void tdm_load_tile_2d(unsigned lds_byte_addr,
                                                 const unsigned short* gptr,
                                                 unsigned tensor_d0,
                                                 unsigned tensor_d1,
                                                 unsigned long long stride0) {
  unsigned long long ga = (unsigned long long)gptr;
  v4u g0;
  g0.x = 1u;                                            // count=1 (valid D#)
  g0.y = lds_byte_addr;                                 // [63:32] lds_addr
  g0.z = (unsigned)(ga & 0xffffffffu);                  // [95:64] global_addr lo
  g0.w = (unsigned)((ga >> 32) & 0x01ffffffu) |         // [120:96] global_addr hi
         (2u << 30);                                    // [127:126] type=2
  v8u g1;
  g1[0] = (1u << 16)      // data_size = 2 bytes
        | (1u << 20)      // pad_enable
        | (4u << 22)      // pad_interval: 32 dwords of data
        | (3u << 25);     // pad_amount: 4 dwords of pad
  g1[1] = (tensor_d0 & 0xffffu) << 16;                  // tensor_dim0 lo16
  g1[2] = (tensor_d0 >> 16) | ((tensor_d1 & 0xffffu) << 16);
  g1[3] = (tensor_d1 >> 16) | (64u << 16);              // tile_dim0 = 64
  g1[4] = 64u;                                          // tile_dim1 = 64
  g1[5] = (unsigned)(stride0 & 0xffffffffu);            // tensor_dim0_stride lo
  g1[6] = (unsigned)(stride0 >> 32);                    // stride hi16, stride1=0
  g1[7] = 0u;
  v4u gz = (v4u)(0u);                                   // groups 2/3: zero (2D)
  asm volatile("tensor_load_to_lds %0, %1, %2, %3"
               :: "s"(g0), "s"(g1), "s"(gz), "s"(gz)
               : "memory");
}

__device__ __forceinline__ unsigned lds_addr_of(const void* p) {
  // Generic pointer to LDS: low 32 bits are the group-segment byte offset.
  return (unsigned)(unsigned long long)p;
}

// A-fragment (16x32 f16, row-major [M][k], chunk base kb): 2 x b128.
#define LOAD_AFRAG(frag, rowptr, kb)                                        \
  do {                                                                      \
    *(uint4*)&(frag).u[0] = *(const uint4*)&(rowptr)[(kb) + half * 8];      \
    *(uint4*)&(frag).u[4] = *(const uint4*)&(rowptr)[(kb) + 16 + half * 8]; \
  } while (0)

// B-fragment (32x16 f16) from transposed storage [n][k]: 2 x b128.
#define LOAD_BFRAG(frag, rowptr, kb)                                        \
  do {                                                                      \
    *(uint4*)&(frag).u[0] = *(const uint4*)&(rowptr)[(kb) + half * 8];      \
    *(uint4*)&(frag).u[4] = *(const uint4*)&(rowptr)[(kb) + 16 + half * 8]; \
  } while (0)

// ---------------------------------------------------------------------------
// Kernel 1: QKV projection.  out[o,n] = sum_c W[o,c] * x[b,c,n]
// ---------------------------------------------------------------------------
__global__ __launch_bounds__(128)
void qkv_gemm_kernel(const float* __restrict__ x, const float* __restrict__ wqkv,
                     unsigned short* __restrict__ Qh, unsigned short* __restrict__ Kh,
                     unsigned short* __restrict__ Vh)
{
  __shared__ __align__(16) unsigned short As[64][40];   // [o_local][k]
  __shared__ __align__(16) unsigned short BsT[64][40];  // [n_local][k] (transposed)
  __shared__ __align__(16) unsigned short St[64][72];   // output transpose staging

  const int t = threadIdx.x;
  const int lane = t & 31, wave = t >> 5;
  const int half = lane >> 4, l16 = lane & 15;
  const int n0 = blockIdx.x * 64;
  const int o0 = blockIdx.y * 64;
  const int b  = blockIdx.z;

  v8f acc[4];
#pragma unroll
  for (int i = 0; i < 4; ++i) acc[i] = zero8();

  for (int kc = 0; kc < CCH; kc += 32) {
    __syncthreads();
    // Stage A row-major and B transposed [n][k]; k-paired u32 stores.
#pragma unroll
    for (int i = 0; i < 8; ++i) {
      int e = t + i * 128;                       // 1024 f16-pairs each
      int ra = e >> 4, ka = (e & 15) * 2;        // A: pairs along k, float2 read
      float2 a2 = *(const float2*)&wqkv[(o0 + ra) * CCH + kc + ka];
      *(unsigned*)&As[ra][ka] = pack_f16x2(a2.x, a2.y);
      int nb = e & 63, kb = (e >> 6) * 2;        // B: pairs along k (two rows of x)
      long long xb = (long long)(b * CCH + kc + kb) * NSP + n0 + nb;
      *(unsigned*)&BsT[nb][kb] = pack_f16x2(x[xb], x[xb + NSP]);
    }
    if (kc + 32 < CCH) {  // warm L2 for next chunk (global_prefetch_b8)
      __builtin_prefetch(&x[(long long)(b * CCH + kc + 32 + (t >> 2)) * NSP + n0 + (t & 3) * 16], 0, 2);
      __builtin_prefetch(&wqkv[(o0 + (t >> 1)) * CCH + kc + 32 + (t & 1) * 16], 0, 2);
    }
    __syncthreads();

    Frag16 af, bf[4];
    LOAD_AFRAG(af, As[wave * 16 + l16], 0);
#pragma unroll
    for (int nt = 0; nt < 4; ++nt) LOAD_BFRAG(bf[nt], BsT[nt * 16 + l16], 0);
#pragma unroll
    for (int nt = 0; nt < 4; ++nt) acc[nt] = wmma_f16(af, bf[nt], acc[nt]);
  }

  __syncthreads();
  const int mat  = o0 >> 8;            // 0=Q 1=K 2=V (tiles never straddle)
  const int head = (o0 & 255) >> 6;
  const float qscale = (mat == 0) ? 0.125f : 1.0f;  // dh^-0.5 folded into Q

  if (mat == 2) {
    // V: stage [d][n], store transposed layout [B,h,dh,N].
#pragma unroll
    for (int nt = 0; nt < 4; ++nt)
#pragma unroll
      for (int v = 0; v < 8; ++v)
        St[wave * 16 + v + 8 * half][nt * 16 + l16] = f2h(acc[nt][v]);
    __syncthreads();
    int d    = t >> 1;
    int nseg = (t & 1) * 32;
    long long base = ((long long)(b * NHEAD + head) * DH + d) * NSP + n0;
#pragma unroll
    for (int i = 0; i < 8; ++i) {
      int n = nseg + i * 4;
      *(unsigned long long*)&Vh[base + n] =
          *(const unsigned long long*)&St[d][n];
    }
  } else {
    // Q/K: stage [n][d], store [B,h,N,dh].
#pragma unroll
    for (int nt = 0; nt < 4; ++nt)
#pragma unroll
      for (int v = 0; v < 8; ++v)
        St[nt * 16 + l16][wave * 16 + v + 8 * half] = f2h(acc[nt][v] * qscale);
    __syncthreads();
    unsigned short* dst = (mat == 0) ? Qh : Kh;
    int n    = t >> 1;
    int dseg = (t & 1) * 32;
    long long base = ((long long)(b * NHEAD + head) * NSP + n0 + n) * DH;
#pragma unroll
    for (int i = 0; i < 8; ++i) {
      int d = dseg + i * 4;
      *(unsigned long long*)&dst[base + d] =
          *(const unsigned long long*)&St[n][d];
    }
  }
}

// ---------------------------------------------------------------------------
// Kernel 2: flash attention with TDM double buffering.
// Block = (b,h, 64-query tile), 4 waves x 16 rows.
// Per key-tile of 64: 8 WMMA for S=Q*K^T, online softmax, 8 WMMA for O+=P*V,
// while the TDM streams the next K/V tiles into the spare LDS buffers.
// ---------------------------------------------------------------------------
__global__ __launch_bounds__(128)
void flash_attn_kernel(const unsigned short* __restrict__ Qh,
                       const unsigned short* __restrict__ Kh,
                       const unsigned short* __restrict__ Vh,   // [B,h,dh,N]
                       unsigned short* __restrict__ Oh)          // [B,N,C]
{
  __shared__ __align__(16) unsigned short Kt[2][64][72];   // [buf][key][d]
  __shared__ __align__(16) unsigned short Vt[2][64][72];   // [buf][d][key]
  __shared__ __align__(16) unsigned short Ps[4][16][72];   // per-wave P [m][key]

  const int t = threadIdx.x;
  const int lane = t & 31, wave = t >> 5;
  const int half = lane >> 4, l16 = lane & 15;
  const int q0 = blockIdx.x * 64;
  const int bh = blockIdx.y;
  const long long mbase = (long long)bh * NSP * DH;  // same for K and V^T

  // Q A-fragments (dh=64 -> two K=32 chunks) as b128 global loads.
  Frag16 qf[2];
  {
    const unsigned short* qrow = &Qh[mbase + (long long)(q0 + wave * 16 + l16) * DH];
#pragma unroll
    for (int kk = 0; kk < 2; ++kk) LOAD_AFRAG(qf[kk], qrow, kk * 32);
  }

  v8f Oacc[4];
#pragma unroll
  for (int i = 0; i < 4; ++i) Oacc[i] = zero8();
  float m[8], l[8];
#pragma unroll
  for (int v = 0; v < 8; ++v) { m[v] = -1e30f; l[v] = 0.0f; }

  // Prologue: DMA tile 0 into buffer 0 (one wave issues; TDM ignores EXEC,
  // so the issue is guarded by a wave-uniform branch).
  if (wave == 0) {
    tdm_load_tile_2d(lds_addr_of(&Kt[0][0][0]), &Kh[mbase], 64u, (unsigned)NSP, 64ull);
    tdm_load_tile_2d(lds_addr_of(&Vt[0][0][0]), &Vh[mbase], (unsigned)NSP, 64u,
                     (unsigned long long)NSP);
  }

  for (int kt = 0; kt < NSP / 64; ++kt) {
    const int cur = kt & 1, nxt = cur ^ 1;
    __syncthreads();  // readers of buf[nxt] (iteration kt-1) are done
    if (wave == 0) {
      if (kt + 1 < NSP / 64) {
        tdm_load_tile_2d(lds_addr_of(&Kt[nxt][0][0]),
                         &Kh[mbase + (long long)(kt + 1) * 64 * DH],
                         64u, (unsigned)NSP, 64ull);
        tdm_load_tile_2d(lds_addr_of(&Vt[nxt][0][0]),
                         &Vh[mbase + (kt + 1) * 64],
                         (unsigned)NSP, 64u, (unsigned long long)NSP);
        __builtin_amdgcn_s_wait_tensorcnt(2);  // pair for tile kt complete
      } else {
        __builtin_amdgcn_s_wait_tensorcnt(0);
      }
    }
    __syncthreads();  // current buffer visible to all waves

    // S = Q * K^T : B[k=d][n=key] -> transposed storage is Kt[key][d].
    v8f S[4];
#pragma unroll
    for (int nt = 0; nt < 4; ++nt) S[nt] = zero8();
#pragma unroll
    for (int kk = 0; kk < 2; ++kk) {
      Frag16 kb[4];
#pragma unroll
      for (int nt = 0; nt < 4; ++nt) LOAD_BFRAG(kb[nt], Kt[cur][nt * 16 + l16], kk * 32);
#pragma unroll
      for (int nt = 0; nt < 4; ++nt) S[nt] = wmma_f16(qf[kk], kb[nt], S[nt]);
    }

    // Online softmax over the 64-wide key tile (half-wave row reductions).
#pragma unroll
    for (int v = 0; v < 8; ++v) {
      float mx = fmaxf(fmaxf(S[0][v], S[1][v]), fmaxf(S[2][v], S[3][v]));
#pragma unroll
      for (int off = 8; off > 0; off >>= 1)
        mx = fmaxf(mx, __shfl_xor(mx, off, 16));
      float mnew = fmaxf(m[v], mx);
      float corr = __expf(m[v] - mnew);
      m[v] = mnew;
      l[v] *= corr;
#pragma unroll
      for (int nt = 0; nt < 4; ++nt) Oacc[nt][v] *= corr;
      float rs = 0.0f;
#pragma unroll
      for (int nt = 0; nt < 4; ++nt) {
        float p = __expf(S[nt][v] - mnew);
        S[nt][v] = p;
        rs += p;
      }
#pragma unroll
      for (int off = 8; off > 0; off >>= 1)
        rs += __shfl_xor(rs, off, 16);
      l[v] += rs;
    }

    // C/D layout -> A layout via per-wave LDS staging.
#pragma unroll
    for (int nt = 0; nt < 4; ++nt)
#pragma unroll
      for (int v = 0; v < 8; ++v)
        Ps[wave][v + 8 * half][nt * 16 + l16] = f2h(S[nt][v]);
    __syncthreads();

    Frag16 pf[2];
#pragma unroll
    for (int kk = 0; kk < 2; ++kk) LOAD_AFRAG(pf[kk], Ps[wave][l16], kk * 32);

    // O += P * V : B[k=key][n=d] -> transposed storage is Vt[d][key].
#pragma unroll
    for (int kk = 0; kk < 2; ++kk) {
      Frag16 vb[4];
#pragma unroll
      for (int nt = 0; nt < 4; ++nt) LOAD_BFRAG(vb[nt], Vt[cur][nt * 16 + l16], kk * 32);
#pragma unroll
      for (int nt = 0; nt < 4; ++nt) Oacc[nt] = wmma_f16(pf[kk], vb[nt], Oacc[nt]);
    }
  }

  // Normalize, stage [n][d] (reuse Kt[0]), store O as f16 [B,N,C].
#pragma unroll
  for (int v = 0; v < 8; ++v) {
    float inv = 1.0f / l[v];
#pragma unroll
    for (int nt = 0; nt < 4; ++nt) Oacc[nt][v] *= inv;
  }
  __syncthreads();
#pragma unroll
  for (int nt = 0; nt < 4; ++nt)
#pragma unroll
    for (int v = 0; v < 8; ++v)
      Kt[0][wave * 16 + v + 8 * half][nt * 16 + l16] = f2h(Oacc[nt][v]);  // [n][d]
  __syncthreads();
  const int b = bh >> 2, head = bh & 3;
  {
    int nl   = t >> 1;
    int dseg = (t & 1) * 32;
    long long base = ((long long)b * NSP + q0 + nl) * CCH + head * DH;
#pragma unroll
    for (int i = 0; i < 8; ++i) {
      int d = dseg + i * 4;
      *(unsigned long long*)&Oh[base + d] =
          *(const unsigned long long*)&Kt[0][nl][d];
    }
  }
}

// ---------------------------------------------------------------------------
// Kernel 3: output projection + bias.  out[b,o,n] = sum_c Wp[o,c]*O[b,n,c] + bias[o]
// ---------------------------------------------------------------------------
__global__ __launch_bounds__(128)
void proj_gemm_kernel(const unsigned short* __restrict__ Oh,  // [B,N,C]
                      const float* __restrict__ wproj,
                      const float* __restrict__ bproj,
                      float* __restrict__ out)
{
  __shared__ __align__(16) unsigned short As[64][40];   // [o_local][k]
  __shared__ __align__(16) unsigned short BsT[64][40];  // [n_local][k]

  const int t = threadIdx.x;
  const int lane = t & 31, wave = t >> 5;
  const int half = lane >> 4, l16 = lane & 15;
  const int n0 = blockIdx.x * 64;
  const int o0 = blockIdx.y * 64;
  const int b  = blockIdx.z;

  v8f acc[4];
#pragma unroll
  for (int i = 0; i < 4; ++i) acc[i] = zero8();

  for (int kc = 0; kc < CCH; kc += 32) {
    __syncthreads();
#pragma unroll
    for (int i = 0; i < 8; ++i) {
      int e = t + i * 128;
      int ra = e >> 4, ka = (e & 15) * 2;        // pairs along k, float2 read
      float2 a2 = *(const float2*)&wproj[(o0 + ra) * CCH + kc + ka];
      *(unsigned*)&As[ra][ka] = pack_f16x2(a2.x, a2.y);
    }
    // B chunk already in [n][c] orientation in memory: b128 copies.
#pragma unroll
    for (int i = 0; i < 2; ++i) {
      int e = t + i * 128;              // 256 uint4
      int row = e >> 2, c8 = (e & 3) * 8;
      *(uint4*)&BsT[row][c8] =
          *(const uint4*)&Oh[((long long)b * NSP + n0 + row) * CCH + kc + c8];
    }
    if (kc + 32 < CCH) {
      __builtin_prefetch(&wproj[(o0 + (t >> 1)) * CCH + kc + 32 + (t & 1) * 16], 0, 2);
      if (t < 64)
        __builtin_prefetch(&Oh[((long long)b * NSP + n0 + t) * CCH + kc + 32], 0, 2);
    }
    __syncthreads();

    Frag16 af, bf[4];
    LOAD_AFRAG(af, As[wave * 16 + l16], 0);
#pragma unroll
    for (int nt = 0; nt < 4; ++nt) LOAD_BFRAG(bf[nt], BsT[nt * 16 + l16], 0);
#pragma unroll
    for (int nt = 0; nt < 4; ++nt) acc[nt] = wmma_f16(af, bf[nt], acc[nt]);
  }

  // f32 epilogue with bias; half-wave writes 64B contiguous runs.
#pragma unroll
  for (int v = 0; v < 8; ++v) {
    int o = o0 + wave * 16 + v + 8 * half;
    float bias = bproj[o];
#pragma unroll
    for (int nt = 0; nt < 4; ++nt) {
      int n = n0 + nt * 16 + l16;
      out[(long long)(b * CCH + o) * NSP + n] = acc[nt][v] + bias;
    }
  }
}

// ---------------------------------------------------------------------------
extern "C" void kernel_launch(void* const* d_in, const int* in_sizes, int n_in,
                              void* d_out, int out_size, void* d_ws, size_t ws_size,
                              hipStream_t stream) {
  (void)in_sizes; (void)n_in; (void)out_size; (void)ws_size;
  const float* x     = (const float*)d_in[0];
  const float* wqkv  = (const float*)d_in[1];
  const float* wproj = (const float*)d_in[2];
  const float* bproj = (const float*)d_in[3];
  float* out = (float*)d_out;

  unsigned short* ws16 = (unsigned short*)d_ws;
  const long long per = (long long)BATCH * NHEAD * NSP * DH;  // 4,194,304 f16
  unsigned short* Qh = ws16;
  unsigned short* Kh = Qh + per;
  unsigned short* Vh = Kh + per;   // transposed [B,h,dh,N]
  unsigned short* Oh = Vh + per;   // [B,N,C]; total scratch 33.6 MB

  qkv_gemm_kernel<<<dim3(NSP / 64, (3 * CCH) / 64, BATCH), 128, 0, stream>>>(
      x, wqkv, Qh, Kh, Vh);
  flash_attn_kernel<<<dim3(NSP / 64, BATCH * NHEAD), 128, 0, stream>>>(
      Qh, Kh, Vh, Oh);
  proj_gemm_kernel<<<dim3(NSP / 64, CCH / 64, BATCH), 128, 0, stream>>>(
      Oh, wproj, bproj, out);
}
